// Processor_71425306133172
// MI455X (gfx1250) — matrix operand order repaired and checked by reading, not compiled, not run on hardware
//
#include <hip/hip_runtime.h>

#define NN 20000
#define NE 320000
#define LD 128
#define SS 4
#define TM 32
#define EPSF 1e-5f

typedef unsigned short u16;
typedef __attribute__((ext_vector_type(16))) __bf16 v16bf;
typedef __attribute__((ext_vector_type(8)))  float  v8f;
typedef __attribute__((ext_vector_type(8)))  unsigned short us8;
typedef __attribute__((ext_vector_type(16))) unsigned short us16;

__device__ static inline u16 f2bf(float f) {
  unsigned u = __builtin_bit_cast(unsigned, f);
  unsigned r = u + 0x7FFFu + ((u >> 16) & 1u);
  return (u16)(r >> 16);
}

// B fragment: weights pre-stored column-major bf16 (wT[col][K]); contiguous k order (same as A),
// lane = col (lane%16), lanes 0-15 hold K=kc*32+0..15, lanes 16-31 hold K=kc*32+16..31,
// i.e. 16 contiguous bf16 = one 32B load per lane.
__device__ static inline v16bf load_b_frag(const u16* colPtr, int kc, int hsel) {
  return __builtin_bit_cast(v16bf, *(const us16*)(colPtr + kc * 32 + hsel * 16));
}

// A fragment from LDS row (contiguous k order, matching B):
// elems 0..7  = A[row][kc*32 + hsel*16     + 0..7]
// elems 8..15 = A[row][kc*32 + hsel*16 + 8 + 0..7]
#define MLP_K_LOOP(AROW, WT, KDIM, ACC0, ACC1)                                        \
  for (int kc = 0; kc < (KDIM) / 32; ++kc) {                                          \
    us8 alo = *(const us8*)((AROW) + kc * 32 + hsel * 16);                            \
    us8 ahi = *(const us8*)((AROW) + kc * 32 + hsel * 16 + 8);                        \
    v16bf af = __builtin_bit_cast(v16bf, __builtin_shufflevector(                     \
        alo, ahi, 0, 1, 2, 3, 4, 5, 6, 7, 8, 9, 10, 11, 12, 13, 14, 15));            \
    v16bf bf0 = load_b_frag((WT) + (size_t)(cb * 32 + lrow) * (KDIM), kc, hsel);      \
    v16bf bf1 = load_b_frag((WT) + (size_t)(cb * 32 + 16 + lrow) * (KDIM), kc, hsel); \
    ACC0 = __builtin_amdgcn_wmma_f32_16x16x32_bf16(false, af, false, bf0,             \
                                                   (short)0, ACC0, false, false);     \
    ACC1 = __builtin_amdgcn_wmma_f32_16x16x32_bf16(false, af, false, bf1,             \
                                                   (short)0, ACC1, false, false);     \
  }

// C/D layout: lane -> col = lane%16; VGPR j -> row = hsel*8 + j (within 16-row tile).
#define STORE_RELU_BF16(DST, BIAS, ACC0, ACC1)                                        \
  {                                                                                   \
    float bb0 = (BIAS)[cb * 32 + lrow];                                               \
    float bb1 = (BIAS)[cb * 32 + 16 + lrow];                                          \
    int orow = rb * 16 + hsel * 8;                                                    \
    _Pragma("unroll")                                                                 \
    for (int j = 0; j < 8; ++j) {                                                     \
      DST[orow + j][cb * 32 + lrow]      = f2bf(fmaxf(ACC0[j] + bb0, 0.f));           \
      DST[orow + j][cb * 32 + 16 + lrow] = f2bf(fmaxf(ACC1[j] + bb1, 0.f));           \
    }                                                                                 \
  }

#define STORE_F32(DST, BIAS, ACC0, ACC1)                                              \
  {                                                                                   \
    float bb0 = (BIAS)[cb * 32 + lrow];                                               \
    float bb1 = (BIAS)[cb * 32 + 16 + lrow];                                          \
    int orow = rb * 16 + hsel * 8;                                                    \
    _Pragma("unroll")                                                                 \
    for (int j = 0; j < 8; ++j) {                                                     \
      DST[orow + j][cb * 32 + lrow]      = ACC0[j] + bb0;                             \
      DST[orow + j][cb * 32 + 16 + lrow] = ACC1[j] + bb1;                             \
    }                                                                                 \
  }

// ---------------- Edge kernel: gather -> 4-layer MLP (WMMA) -> LN -> residual+scatter
__global__ __launch_bounds__(256) void edge_kernel(
    const float* __restrict__ xbuf, float* __restrict__ eattr,
    const int* __restrict__ ei,
    const u16* __restrict__ w0T, const u16* __restrict__ w1T,
    const u16* __restrict__ w2T, const u16* __restrict__ w3T,
    const float* __restrict__ b0, const float* __restrict__ b1,
    const float* __restrict__ b2, const float* __restrict__ b3,
    const float* __restrict__ gam, const float* __restrict__ bet,
    float* __restrict__ agg) {
  __shared__ __attribute__((aligned(32))) u16 sIn[TM][392];   // 3L input, padded
  __shared__ __attribute__((aligned(32))) u16 sAct[TM][136];  // inter-layer act, padded
  __shared__ float sOut[TM][132];                             // final pre-LN, padded

  const int tid  = threadIdx.x;
  const int lane = tid & 31;
  const int wv   = tid >> 5;
  const int rb   = wv & 1;   // row block (16 rows)
  const int cb   = wv >> 1;  // col group (32 cols)
  const int lrow = lane & 15;
  const int hsel = lane >> 4;
  const int e0   = blockIdx.x * TM;

  // Gather [x[recv] | x[send] | edge_attr] -> bf16 in LDS (48 elems/thread, uniform)
  for (int i = tid; i < TM * 384; i += 256) {
    int r = i / 384, c = i - r * 384;
    int e = e0 + r;
    float v;
    if (c < 128)      v = xbuf[(size_t)ei[NE + e] * LD + c];         // recv
    else if (c < 256) v = xbuf[(size_t)ei[e] * LD + (c - 128)];      // send
    else              v = eattr[(size_t)e * LD + (c - 256)];
    sIn[r][c] = f2bf(v);
  }
  __syncthreads();

  const u16* arow0 = &sIn[rb * 16 + lrow][0];
  const u16* arow  = &sAct[rb * 16 + lrow][0];

  v8f l0a = {}, l0b = {};
  MLP_K_LOOP(arow0, w0T, 384, l0a, l0b);
  __syncthreads();
  STORE_RELU_BF16(sAct, b0, l0a, l0b);
  __syncthreads();

  v8f l1a = {}, l1b = {};
  MLP_K_LOOP(arow, w1T, 128, l1a, l1b);
  __syncthreads();
  STORE_RELU_BF16(sAct, b1, l1a, l1b);
  __syncthreads();

  v8f l2a = {}, l2b = {};
  MLP_K_LOOP(arow, w2T, 128, l2a, l2b);
  __syncthreads();
  STORE_RELU_BF16(sAct, b2, l2a, l2b);
  __syncthreads();

  v8f l3a = {}, l3b = {};
  MLP_K_LOOP(arow, w3T, 128, l3a, l3b);
  __syncthreads();
  STORE_F32(sOut, b3, l3a, l3b);
  __syncthreads();

  // LayerNorm: 8 lanes per row, 16 cols each; reduce via wave32 shfl_xor.
  const int r  = tid >> 3;
  const int c0 = (tid & 7) * 16;
  float s = 0.f;
#pragma unroll
  for (int j = 0; j < 16; ++j) s += sOut[r][c0 + j];
  s += __shfl_xor(s, 1); s += __shfl_xor(s, 2); s += __shfl_xor(s, 4);
  float mu = s * (1.0f / LD);
  float vv = 0.f;
#pragma unroll
  for (int j = 0; j < 16; ++j) { float d = sOut[r][c0 + j] - mu; vv += d * d; }
  vv += __shfl_xor(vv, 1); vv += __shfl_xor(vv, 2); vv += __shfl_xor(vv, 4);
  float rstd = rsqrtf(vv * (1.0f / LD) + EPSF);
  int e = e0 + r;
  int dst = ei[NE + e];  // recv (segment_sum target)
#pragma unroll
  for (int j = 0; j < 16; ++j) {
    int c = c0 + j;
    float m = (sOut[r][c] - mu) * rstd * gam[c] + bet[c];
    eattr[(size_t)e * LD + c] += m;                  // edge residual
    atomicAdd(&agg[(size_t)dst * LD + c], m);        // scatter-add
  }
}

// ---------------- Node kernel: [x | agg] -> 4-layer MLP (WMMA) -> LN -> x residual
__global__ __launch_bounds__(256) void node_kernel(
    float* __restrict__ xbuf, const float* __restrict__ agg,
    const u16* __restrict__ w0T, const u16* __restrict__ w1T,
    const u16* __restrict__ w2T, const u16* __restrict__ w3T,
    const float* __restrict__ b0, const float* __restrict__ b1,
    const float* __restrict__ b2, const float* __restrict__ b3,
    const float* __restrict__ gam, const float* __restrict__ bet) {
  __shared__ __attribute__((aligned(32))) u16 sIn[TM][264];  // 2L input, padded
  __shared__ __attribute__((aligned(32))) u16 sAct[TM][136];
  __shared__ float sOut[TM][132];

  const int tid  = threadIdx.x;
  const int lane = tid & 31;
  const int wv   = tid >> 5;
  const int rb   = wv & 1;
  const int cb   = wv >> 1;
  const int lrow = lane & 15;
  const int hsel = lane >> 4;
  const int n0   = blockIdx.x * TM;

  for (int i = tid; i < TM * 256; i += 256) {
    int r = i >> 8, c = i & 255;
    int nd = n0 + r;
    float v = (c < 128) ? xbuf[(size_t)nd * LD + c]
                        : agg[(size_t)nd * LD + (c - 128)];
    sIn[r][c] = f2bf(v);
  }
  __syncthreads();

  const u16* arow0 = &sIn[rb * 16 + lrow][0];
  const u16* arow  = &sAct[rb * 16 + lrow][0];

  v8f l0a = {}, l0b = {};
  MLP_K_LOOP(arow0, w0T, 256, l0a, l0b);
  __syncthreads();
  STORE_RELU_BF16(sAct, b0, l0a, l0b);
  __syncthreads();

  v8f l1a = {}, l1b = {};
  MLP_K_LOOP(arow, w1T, 128, l1a, l1b);
  __syncthreads();
  STORE_RELU_BF16(sAct, b1, l1a, l1b);
  __syncthreads();

  v8f l2a = {}, l2b = {};
  MLP_K_LOOP(arow, w2T, 128, l2a, l2b);
  __syncthreads();
  STORE_RELU_BF16(sAct, b2, l2a, l2b);
  __syncthreads();

  v8f l3a = {}, l3b = {};
  MLP_K_LOOP(arow, w3T, 128, l3a, l3b);
  __syncthreads();
  STORE_F32(sOut, b3, l3a, l3b);
  __syncthreads();

  const int r  = tid >> 3;
  const int c0 = (tid & 7) * 16;
  float s = 0.f;
#pragma unroll
  for (int j = 0; j < 16; ++j) s += sOut[r][c0 + j];
  s += __shfl_xor(s, 1); s += __shfl_xor(s, 2); s += __shfl_xor(s, 4);
  float mu = s * (1.0f / LD);
  float vv = 0.f;
#pragma unroll
  for (int j = 0; j < 16; ++j) { float d = sOut[r][c0 + j] - mu; vv += d * d; }
  vv += __shfl_xor(vv, 1); vv += __shfl_xor(vv, 2); vv += __shfl_xor(vv, 4);
  float rstd = rsqrtf(vv * (1.0f / LD) + EPSF);
  int nd = n0 + r;
#pragma unroll
  for (int j = 0; j < 16; ++j) {
    int c = c0 + j;
    float m = (sOut[r][c] - mu) * rstd * gam[c] + bet[c];
    xbuf[(size_t)nd * LD + c] += m;  // node residual
  }
}

// Weight prep: [S][K][128] f32 -> [S][128][K] bf16 (column-major for B fragments)
__global__ __launch_bounds__(256) void prep_w(const float* __restrict__ src,
                                              u16* __restrict__ dst, int K) {
  int idx = blockIdx.x * 256 + threadIdx.x;
  int total = SS * K * LD;
  if (idx >= total) return;
  int s = idx / (K * LD);
  int rem = idx - s * (K * LD);
  int k = rem / LD;
  int c = rem - k * LD;
  dst[(size_t)s * LD * K + (size_t)c * K + k] = f2bf(src[idx]);
}

__global__ __launch_bounds__(256) void zero_f(float* __restrict__ p, int n) {
  int i = blockIdx.x * 256 + threadIdx.x;
  if (i < n) p[i] = 0.f;
}

extern "C" void kernel_launch(void* const* d_in, const int* in_sizes, int n_in,
                              void* d_out, int out_size, void* d_ws, size_t ws_size,
                              hipStream_t stream) {
  (void)in_sizes; (void)n_in; (void)out_size; (void)ws_size;
  const float* x  = (const float*)d_in[0];
  const float* ea = (const float*)d_in[1];
  const int*   ei = (const int*)d_in[2];
  const float* e_w[4] = {(const float*)d_in[3], (const float*)d_in[5],
                         (const float*)d_in[7], (const float*)d_in[9]};
  const float* e_b[4] = {(const float*)d_in[4], (const float*)d_in[6],
                         (const float*)d_in[8], (const float*)d_in[10]};
  const float* e_g = (const float*)d_in[11];
  const float* e_be = (const float*)d_in[12];
  const float* n_w[4] = {(const float*)d_in[13], (const float*)d_in[15],
                         (const float*)d_in[17], (const float*)d_in[19]};
  const float* n_b[4] = {(const float*)d_in[14], (const float*)d_in[16],
                         (const float*)d_in[18], (const float*)d_in[20]};
  const float* n_g = (const float*)d_in[21];
  const float* n_be = (const float*)d_in[22];

  float* outx = (float*)d_out;                 // [N][128] live x
  float* oute = outx + (size_t)NN * LD;        // [E][128] live edge_attr
  float* agg  = (float*)d_ws;                  // [N][128]
  u16* wb = (u16*)((char*)d_ws + (size_t)NN * LD * sizeof(float));
  u16* eW0T = wb;
  u16* eW1T = eW0T + (size_t)SS * LD * 384;
  u16* eW2T = eW1T + (size_t)SS * LD * 128;
  u16* eW3T = eW2T + (size_t)SS * LD * 128;
  u16* nW0T = eW3T + (size_t)SS * LD * 128;
  u16* nW1T = nW0T + (size_t)SS * LD * 256;
  u16* nW2T = nW1T + (size_t)SS * LD * 128;
  u16* nW3T = nW2T + (size_t)SS * LD * 128;

  hipMemcpyAsync(outx, x, (size_t)NN * LD * sizeof(float),
                 hipMemcpyDeviceToDevice, stream);
  hipMemcpyAsync(oute, ea, (size_t)NE * LD * sizeof(float),
                 hipMemcpyDeviceToDevice, stream);

  const int Ks_e[4] = {384, 128, 128, 128};
  u16* dst_e[4] = {eW0T, eW1T, eW2T, eW3T};
  const int Ks_n[4] = {256, 128, 128, 128};
  u16* dst_n[4] = {nW0T, nW1T, nW2T, nW3T};
  for (int l = 0; l < 4; ++l) {
    int te = SS * Ks_e[l] * LD;
    prep_w<<<(te + 255) / 256, 256, 0, stream>>>(e_w[l], dst_e[l], Ks_e[l]);
    int tn = SS * Ks_n[l] * LD;
    prep_w<<<(tn + 255) / 256, 256, 0, stream>>>(n_w[l], dst_n[l], Ks_n[l]);
  }

  for (int s = 0; s < SS; ++s) {
    zero_f<<<(NN * LD + 255) / 256, 256, 0, stream>>>(agg, NN * LD);
    edge_kernel<<<NE / TM, 256, 0, stream>>>(
        outx, oute, ei,
        eW0T + (size_t)s * LD * 384, eW1T + (size_t)s * LD * 128,
        eW2T + (size_t)s * LD * 128, eW3T + (size_t)s * LD * 128,
        e_b[0] + s * LD, e_b[1] + s * LD, e_b[2] + s * LD, e_b[3] + s * LD,
        e_g + s * LD, e_be + s * LD, agg);
    node_kernel<<<NN / TM, 256, 0, stream>>>(
        outx, agg,
        nW0T + (size_t)s * LD * 256, nW1T + (size_t)s * LD * 128,
        nW2T + (size_t)s * LD * 128, nW3T + (size_t)s * LD * 128,
        n_b[0] + s * LD, n_b[1] + s * LD, n_b[2] + s * LD, n_b[3] + s * LD,
        n_g + s * LD, n_be + s * LD);
  }
}
